// DistributedAFNONet_824633720962
// MI455X (gfx1250) — compile-verified, matrix operand order checked by reading, not run on hardware
//
#include <hip/hip_runtime.h>
#include <hip/hip_bf16.h>
#include <math.h>

// ---------------------------------------------------------------------------
// AFNO block for MI455X / gfx1250 (wave32, WMMA).
//   out = x + irfft2( softshrink( blockMLP( rfft2(x)[band] ) ) ), ortho norm.
// Bandwidth-bound (~2-3 GB HBM traffic at 23.3 TB/s). The ~50-GFLOP complex
// block-MLP runs on v_wmma_f32_16x16x32_f16 with all fragments loaded as
// contiguous 32B chunks (ds_load_b128 / global_load_b128), weights pre-packed
// into WMMA B-fragment order by a one-shot kernel.
// ---------------------------------------------------------------------------

typedef __attribute__((ext_vector_type(16))) _Float16 v16h;
typedef __attribute__((ext_vector_type(8)))  float    v8f;

#define BATCH   2
#define CHAN    768
#define NBLK    8
#define BS      96          // channels per block
#define HD      256
#define WC      103         // kept_modes = int(129*0.8)
#define LO      26
#define HI      232
#define BAND_H  206
#define PPB     (BAND_H*WC) // 21218
#define PTOT    (BATCH*PPB) // 42436
#define NTILES  ((PTOT+15)/16)  // 2653
#define TPW     8           // M-tiles (waves) per workgroup
#define LAM     0.01f
#define ORTHO   0.0625f     // 1/sqrt(256) per transform dim

// Packed-weight geometry: [k][layer][path][n][s][lane][h] f16
#define FRAG_H     512                 // 32 lanes * 16 halves
#define WF_PER_K   (2*2*6*6*FRAG_H)    // 73728 halves per block k
#define WF_PATH    (6*6*FRAG_H)        // path stride  = 18432
#define WF_LAYER   (2*6*6*FRAG_H)      // layer stride = 36864
#define F_ELEMS    ((size_t)BATCH*CHAN*HD*WC)   // float2 elements in spectrum ws

__device__ __forceinline__ int brev8(int i) { return __brev(i) >> 24; }

// Map matrix element (K, M-or-N) to WMMA 16-bit fragment slot (s, lane, h).
__device__ __forceinline__ void frag_pos(int K, int M, int& s, int& lane, int& h) {
  s = K >> 5;
  int kk = K & 31, lh, hh;
  if (kk < 8)       { lh = 0; hh = kk; }
  else if (kk < 16) { lh = 1; hh = kk - 8; }
  else if (kk < 24) { lh = 0; hh = kk - 8; }
  else              { lh = 1; hh = kk - 16; }
  lane = (lh << 4) | M;
  h = hh;
}

// 256-point in-LDS complex FFT, 128 threads, input pre-bit-reversed.
__device__ void fft256(float* re, float* im, int tid, float sgn) {
  #pragma unroll
  for (int s = 1; s <= 8; ++s) {
    int half = 1 << (s - 1);
    int pos  = tid & (half - 1);
    int base = (tid >> (s - 1)) << s;
    int i0 = base + pos, i1 = i0 + half;
    float ang = sgn * 6.28318530717958647f * (float)pos / (float)(1 << s);
    float c, sn;
    __sincosf(ang, &sn, &c);
    float ar = re[i1], ai = im[i1];
    float tr = c * ar - sn * ai;
    float ti = c * ai + sn * ar;
    float br = re[i0], bi = im[i0];
    re[i1] = br - tr; im[i1] = bi - ti;
    re[i0] = br + tr; im[i0] = bi + ti;
    __syncthreads();
  }
}

// ---- Weight pre-pack: combined complex weights -> B-fragment order ---------
// real path: [Wr ; -Wi], imag path: [Wi ; Wr]  (K = 192 concat dimension)
__global__ void k_pack_weights(const float* __restrict__ w1,
                               const float* __restrict__ w2,
                               _Float16* __restrict__ Wf) {
  int k = blockIdx.x;
  int tid = threadIdx.x;
  for (int e = tid; e < WF_PER_K; e += 256) {
    int h    = e & 15;
    int lane = (e >> 4) & 31;
    int t    = e >> 9;            // frag index
    int s    = t % 6;  t /= 6;
    int n    = t % 6;  t /= 6;
    int path = t & 1;
    int layer = t >> 1;
    int kb = (lane >> 4) << 3;
    int kk = (h < 8) ? (h + kb) : (h + 8 + kb);
    int K = s * 32 + kk;
    int N = n * 16 + (lane & 15);
    const float* W = (layer == 0 ? w1 : w2) + (size_t)k * BS * BS * 2;
    float val;
    if (K < BS) val = W[(K * BS + N) * 2 + path];
    else {
      int Km = K - BS;
      val = (path == 0) ? -W[(Km * BS + N) * 2 + 1] : W[(Km * BS + N) * 2 + 0];
    }
    Wf[(size_t)k * WF_PER_K + e] = (_Float16)val;
  }
}

// ---- Stage 1: rfft along W; keep bins [0,103). -----------------------------
__global__ void k_row_fft_fwd(const float* __restrict__ x, float2* __restrict__ F) {
  __shared__ float re[256], im[256];
  int tid = threadIdx.x;
  size_t row = blockIdx.x;                       // img*256 + h
  const float* xr = x + row * 256;
  for (int i = tid; i < 256; i += 128) {
    int j = brev8(i);
    re[j] = xr[i];
    im[j] = 0.f;
  }
  __syncthreads();
  fft256(re, im, tid, -1.f);
  float2* out = F + row * WC;
  if (tid < WC) out[tid] = make_float2(re[tid] * ORTHO, im[tid] * ORTHO);
}

// ---- Stage 2: complex FFT along H, only the 103 kept columns. --------------
__global__ void k_col_fft_fwd(float2* __restrict__ F) {
  __shared__ float re[256], im[256];
  int tid = threadIdx.x;
  int wp = blockIdx.x % WC;
  size_t img = blockIdx.x / WC;
  float2* col = F + img * (size_t)(HD * WC) + wp;
  for (int i = tid; i < 256; i += 128) {
    float2 v = col[(size_t)i * WC];
    int j = brev8(i);
    re[j] = v.x; im[j] = v.y;
  }
  __syncthreads();
  fft256(re, im, tid, -1.f);
  for (int i = tid; i < 256; i += 128)
    col[(size_t)i * WC] = make_float2(re[i] * ORTHO, im[i] * ORTHO);
}

// ---- Stage 3: per-block complex MLP on the band, via WMMA ------------------
__global__ void k_afno_mlp(float2* __restrict__ F,
                           const _Float16* __restrict__ Wf,
                           const float* __restrict__ b1,
                           const float* __restrict__ b2) {
  extern __shared__ _Float16 smem[];
  // Fragment-ordered activation buffers: per wave [s(6)][lane(32)][h(16)]
  _Float16* bufA  = smem;                 // 8 waves * 3072 halves
  _Float16* bufO1 = bufA + TPW * 6 * FRAG_H;

  int tid = threadIdx.x;
  int k   = blockIdx.y;
  int tile0 = blockIdx.x * TPW;

  // Stage activations: 128 band pixels x 96 complex channels, scattered into
  // WMMA A-fragment slots (coalesced 8B global loads, 16-bit LDS stores).
  for (int idx = tid; idx < TPW * 16 * BS; idx += 256) {
    int pl = idx & 127, ch = idx >> 7;           // point-local, channel
    int wv = pl >> 4;
    int tile = tile0 + wv;
    if (tile >= NTILES) tile = NTILES - 1;       // clamp; dup work is identical
    int p = tile * 16 + (pl & 15);
    float2 v = make_float2(0.f, 0.f);
    if (p < PTOT) {
      int b = p / PPB, r = p - b * PPB;
      int hx = r / WC, y = r - hx * WC;
      int h = LO + hx;
      v = F[(((size_t)(b * CHAN + k * BS + ch)) * HD + h) * WC + y];
    }
    int M = pl & 15, s, lane, h;
    _Float16* wbuf = bufA + wv * 6 * FRAG_H;
    frag_pos(ch, M, s, lane, h);                 // Ar at K=ch
    wbuf[s * FRAG_H + lane * 16 + h] = (_Float16)v.x;
    frag_pos(ch + BS, M, s, lane, h);            // Ai at K=ch+96
    wbuf[s * FRAG_H + lane * 16 + h] = (_Float16)v.y;
  }
  __syncthreads();

  int wave = tid >> 5, lane = tid & 31;
  int myTile = tile0 + wave;
  if (myTile >= NTILES) myTile = NTILES - 1;
  _Float16* Af  = bufA  + wave * 6 * FRAG_H;
  _Float16* O1f = bufO1 + wave * 6 * FRAG_H;
  const _Float16* Wk = Wf + (size_t)k * WF_PER_K;
  int nloc = lane & 15;
  int kb   = (lane >> 4) << 3;                   // M-offset in D layout

  // ---- Layer 1: o1 = crelu(A @ W1 + b1) -> LDS fragment buffer ----
  for (int n = 0; n < 6; ++n) {
    v8f dr = {0.f, 0.f, 0.f, 0.f, 0.f, 0.f, 0.f, 0.f};
    v8f di = {0.f, 0.f, 0.f, 0.f, 0.f, 0.f, 0.f, 0.f};
    for (int s = 0; s < 6; ++s) {
      v16h a  = *(const v16h*)(Af + s * FRAG_H + lane * 16);
      const _Float16* wb = Wk + (size_t)(n * 6 + s) * FRAG_H + lane * 16; // layer0,path0
      v16h br = *(const v16h*)wb;
      v16h bi = *(const v16h*)(wb + WF_PATH);
      dr = __builtin_amdgcn_wmma_f32_16x16x32_f16(false, a, false, br, (short)0, dr, false, false);
      di = __builtin_amdgcn_wmma_f32_16x16x32_f16(false, a, false, bi, (short)0, di, false, false);
    }
    int N = n * 16 + nloc;
    float2 bb = ((const float2*)b1)[k * BS + N];
    #pragma unroll
    for (int r = 0; r < 8; ++r) {
      int M = r + kb;
      float orr = fmaxf(dr[r] + bb.x, 0.f);
      float oii = fmaxf(di[r] + bb.y, 0.f);
      int s2, l2, h2;
      frag_pos(N, M, s2, l2, h2);
      O1f[s2 * FRAG_H + l2 * 16 + h2] = (_Float16)orr;
      frag_pos(N + BS, M, s2, l2, h2);
      O1f[s2 * FRAG_H + l2 * 16 + h2] = (_Float16)oii;
    }
  }
  __syncthreads();

  // ---- Layer 2: o2 = softshrink(O1 @ W2 + b2) -> band (in-place) ----
  for (int n = 0; n < 6; ++n) {
    v8f dr = {0.f, 0.f, 0.f, 0.f, 0.f, 0.f, 0.f, 0.f};
    v8f di = {0.f, 0.f, 0.f, 0.f, 0.f, 0.f, 0.f, 0.f};
    for (int s = 0; s < 6; ++s) {
      v16h a  = *(const v16h*)(O1f + s * FRAG_H + lane * 16);
      const _Float16* wb = Wk + WF_LAYER + (size_t)(n * 6 + s) * FRAG_H + lane * 16;
      v16h br = *(const v16h*)wb;
      v16h bi = *(const v16h*)(wb + WF_PATH);
      dr = __builtin_amdgcn_wmma_f32_16x16x32_f16(false, a, false, br, (short)0, dr, false, false);
      di = __builtin_amdgcn_wmma_f32_16x16x32_f16(false, a, false, bi, (short)0, di, false, false);
    }
    int N = n * 16 + nloc;
    float2 bb = ((const float2*)b2)[k * BS + N];
    #pragma unroll
    for (int r = 0; r < 8; ++r) {
      int M = r + kb;
      int p = myTile * 16 + M;
      if (p < PTOT) {
        float xr = dr[r] + bb.x, xi = di[r] + bb.y;
        float sr = copysignf(fmaxf(fabsf(xr) - LAM, 0.f), xr);
        float si = copysignf(fmaxf(fabsf(xi) - LAM, 0.f), xi);
        int b = p / PPB, rr = p - b * PPB;
        int hx = rr / WC, y = rr - hx * WC;
        int h = LO + hx;
        F[(((size_t)(b * CHAN + k * BS + N)) * HD + h) * WC + y] = make_float2(sr, si);
      }
    }
  }
}

// ---- Stage 4: inverse column FFT (zero outside band rows). -----------------
__global__ void k_col_fft_inv(float2* __restrict__ F) {
  __shared__ float re[256], im[256];
  int tid = threadIdx.x;
  int wp = blockIdx.x % WC;
  size_t img = blockIdx.x / WC;
  float2* col = F + img * (size_t)(HD * WC) + wp;
  for (int i = tid; i < 256; i += 128) {
    float2 v = (i >= LO && i < HI) ? col[(size_t)i * WC] : make_float2(0.f, 0.f);
    int j = brev8(i);
    re[j] = v.x; im[j] = v.y;
  }
  __syncthreads();
  fft256(re, im, tid, +1.f);
  for (int i = tid; i < 256; i += 128)
    col[(size_t)i * WC] = make_float2(re[i] * ORTHO, im[i] * ORTHO);
}

// ---- Stage 5: inverse row irfft (Hermitian from 103 bins) + residual. ------
__global__ void k_row_ifft_res(const float2* __restrict__ F,
                               const float* __restrict__ x,
                               float* __restrict__ out) {
  __shared__ float re[256], im[256];
  int tid = threadIdx.x;
  size_t row = blockIdx.x;
  const float2* Fr = F + row * WC;
  for (int i = tid; i < 256; i += 128) {
    float2 v = make_float2(0.f, 0.f);
    if (i < WC) v = Fr[i];
    else if (i >= 256 - (WC - 1)) {              // conj(X[256-i])
      float2 u = Fr[256 - i];
      v = make_float2(u.x, -u.y);
    }
    int j = brev8(i);
    re[j] = v.x; im[j] = v.y;
  }
  __syncthreads();
  fft256(re, im, tid, +1.f);
  const float* xr = x + row * 256;
  float* orow = out + row * 256;
  for (int i = tid; i < 256; i += 128)
    orow[i] = re[i] * ORTHO + xr[i];
}

extern "C" void kernel_launch(void* const* d_in, const int* in_sizes, int n_in,
                              void* d_out, int out_size, void* d_ws, size_t ws_size,
                              hipStream_t stream) {
  const float* x  = (const float*)d_in[0];
  const float* w1 = (const float*)d_in[1];
  const float* b1 = (const float*)d_in[2];
  const float* w2 = (const float*)d_in[3];
  const float* b2 = (const float*)d_in[4];
  float* out = (float*)d_out;

  // Workspace: spectrum [B*C][256][103] float2 (324 MB, in-place), then
  // fragment-packed weights (1.18 MB, 32B-aligned).
  float2*   F  = (float2*)d_ws;
  _Float16* Wf = (_Float16*)((char*)d_ws + F_ELEMS * sizeof(float2));

  k_pack_weights<<<NBLK, 256, 0, stream>>>(w1, w2, Wf);

  int rows = BATCH * CHAN * HD;                  // 393216
  k_row_fft_fwd<<<rows, 128, 0, stream>>>(x, F);
  k_col_fft_fwd<<<BATCH * CHAN * WC, 128, 0, stream>>>(F);

  size_t lds = (size_t)(2 * TPW * 6 * FRAG_H) * sizeof(_Float16);   // 96 KB
  static int attr_set = 0;  // idempotent host-side attr; not a stream op
  if (!attr_set) {
    hipFuncSetAttribute((const void*)k_afno_mlp,
                        hipFuncAttributeMaxDynamicSharedMemorySize, (int)lds);
    attr_set = 1;
  }
  dim3 gmlp((NTILES + TPW - 1) / TPW, NBLK);
  k_afno_mlp<<<gmlp, 256, lds, stream>>>(F, Wf, b1, b2);

  k_col_fft_inv<<<BATCH * CHAN * WC, 128, 0, stream>>>(F);
  k_row_ifft_res<<<rows, 128, 0, stream>>>(F, x, out);
}
